// TopDownSceneCaptionModule_53558242181153
// MI455X (gfx1250) — matrix operand order repaired and checked by reading, not compiled, not run on hardware
//
#include <hip/hip_runtime.h>
#include <hip/hip_bf16.h>

// ---------------- problem constants ----------------
#define BB   128
#define TT   30
#define EMB  300
#define FEAT 128
#define HID  512
#define PP   256
#define VV   3433

// padded dims
#define TD_K   960   // 300+512+128=940 -> 960
#define TD_N   304   // 300 -> 304
#define TD_LD  320   // td / lang bf16 buffer leading dim (K for next GEMM)
#define LANG_K 640   // 128+512
#define CLS_N  3440  // 3433 -> 3440
#define GRU_N  1536  // 3*HID

typedef __attribute__((ext_vector_type(16))) __bf16          v16bf;
typedef __attribute__((ext_vector_type(16))) unsigned short  v16u;
typedef __attribute__((ext_vector_type(8)))  unsigned short  v8u;
typedef __attribute__((ext_vector_type(8)))  float           v8f;

__device__ __forceinline__ unsigned short f2bf(float f) {
    union { float f; unsigned int u; } x; x.f = f;
    unsigned int u = x.u;
    unsigned int r = (u + 0x7FFFu + ((u >> 16) & 1u)) >> 16;   // RNE
    return (unsigned short)r;
}

// ---------------- weight convert + pad (f32 [N,K] -> bf16 [Npad,Kpad]) ----------------
__global__ void pad_bf16_kernel(const float* __restrict__ src, int N, int K,
                                unsigned short* __restrict__ dst, int Npad, int Kpad) {
    int idx = blockIdx.x * blockDim.x + threadIdx.x;
    int total = Npad * Kpad;
    if (idx >= total) return;
    int n = idx / Kpad, k = idx - n * Kpad;
    float v = (n < N && k < K) ? src[(size_t)n * K + k] : 0.0f;
    dst[idx] = f2bf(v);
}

__global__ void zero_u32_kernel(unsigned int* __restrict__ p, int n) {
    int idx = blockIdx.x * blockDim.x + threadIdx.x;
    if (idx < n) p[idx] = 0u;
}

// ---------------- WMMA GEMM: C[M,N] = act(A[M,K] * B[N,K]^T + bias) ----------------
// A and B are bf16 (ushort) with padded K. Each wave computes a 16x64 slab
// (4 accumulators reusing one A fragment). Block = 128 threads (4 waves).
// Inner loop is branch-free: out-of-range tiles have their B pointer clamped
// to row 0 (safe reads); only the stores are guarded.
__global__ __launch_bounds__(128)
void wmma_gemm_bf16(const unsigned short* __restrict__ Abf, int lda,
                    const unsigned short* __restrict__ Bw, int ldb,
                    const float* __restrict__ bias,
                    float* __restrict__ C, int ldc,
                    unsigned short* __restrict__ Cbf, int ldcb,
                    int M, int Nreal, int Npad, int Kpad, int act) {
    const int lane = threadIdx.x & 31;
    const int wave = threadIdx.x >> 5;
    const int nb = (blockIdx.x * 4 + wave) * 64;      // 4 tiles of 16 cols
    const int m0 = blockIdx.y * 16;
    if (nb >= Npad) return;

    const int g = lane >> 4;     // half-wave select
    const int r = lane & 15;     // row (A/M) or col (B/N) within tile

    v8f acc[4] = {{}, {}, {}, {}};
    const unsigned short* brow[4];
    #pragma unroll
    for (int tt = 0; tt < 4; ++tt) {
        int n0 = nb + 16 * tt;
        brow[tt] = Bw + (size_t)((n0 < Npad ? n0 : 0) + r) * ldb;
    }
    const unsigned short* arow = Abf + (size_t)(m0 + r) * lda;

    for (int k0 = 0; k0 < Kpad; k0 += 32) {
        // A fragment: elems 0..7 -> K = k0+8g.. ; elems 8..15 -> K = k0+16+8g..
        v8u a0 = *(const v8u*)(arow + k0 + 8 * g);
        v8u a1 = *(const v8u*)(arow + k0 + 16 + 8 * g);
        v16u au = __builtin_shufflevector(a0, a1, 0, 1, 2, 3, 4, 5, 6, 7,
                                                  8, 9, 10, 11, 12, 13, 14, 15);
        v16bf af = __builtin_bit_cast(v16bf, au);
        #pragma unroll
        for (int tt = 0; tt < 4; ++tt) {
            // B fragment: elems 0..7 -> K = k0+8g.. ; elems 8..15 -> K = k0+16+8g..
            v8u b0 = *(const v8u*)(brow[tt] + k0 + 8 * g);
            v8u b1 = *(const v8u*)(brow[tt] + k0 + 16 + 8 * g);
            v16u bu = __builtin_shufflevector(b0, b1, 0, 1, 2, 3, 4, 5, 6, 7,
                                                      8, 9, 10, 11, 12, 13, 14, 15);
            v16bf bfv = __builtin_bit_cast(v16bf, bu);
            acc[tt] = __builtin_amdgcn_wmma_f32_16x16x32_bf16(
                false, af, false, bfv, (short)0, acc[tt], false, false);
        }
    }

    // store: D VGPR j -> row m0 + j + 8g, col n + r  (only real columns)
    #pragma unroll
    for (int tt = 0; tt < 4; ++tt) {
        int col = nb + 16 * tt + r;
        if (col >= Nreal) continue;
        float bv = bias ? bias[col] : 0.0f;
        #pragma unroll
        for (int j = 0; j < 8; ++j) {
            int row = m0 + j + 8 * g;
            float v = acc[tt][j] + bv;
            if (act == 1) v = fmaxf(v, 0.0f);
            if (C)   C[(size_t)row * ldc + col] = v;
            if (Cbf) Cbf[(size_t)row * ldcb + col] = f2bf(v);
        }
    }
}

// ---------------- concat [x_t, h2, target_feat] -> td_in_bf [B, TD_K] ----------------
__global__ void concat_td_kernel(const float* __restrict__ word_embs, int t,
                                 const float* __restrict__ h2,
                                 const float* __restrict__ tfeat,
                                 unsigned short* __restrict__ td_in_bf) {
    int idx = blockIdx.x * blockDim.x + threadIdx.x;
    if (idx >= BB * 940) return;
    int b = idx / 940, c = idx - b * 940;
    float v;
    if (c < EMB)            v = word_embs[((size_t)b * TT + t) * EMB + c];
    else if (c < EMB + HID) v = h2[(size_t)b * HID + (c - EMB)];
    else                    v = tfeat[(size_t)b * FEAT + (c - EMB - HID)];
    td_in_bf[(size_t)b * TD_K + c] = f2bf(v);
}

// ---------------- fused GRU gates: h = (1-z)*n + z*h  (f32 state + bf16 shadow) ----
__global__ void gru_gate_kernel(const float* __restrict__ gi,
                                const float* __restrict__ gh,
                                float* __restrict__ h,
                                unsigned short* __restrict__ h_bf) {
    int idx = blockIdx.x * blockDim.x + threadIdx.x;
    if (idx >= BB * HID) return;
    int b = idx >> 9, j = idx & (HID - 1);
    const float* gib = gi + (size_t)b * GRU_N;
    const float* ghb = gh + (size_t)b * GRU_N;
    float r = 1.0f / (1.0f + __expf(-(gib[j] + ghb[j])));
    float z = 1.0f / (1.0f + __expf(-(gib[HID + j] + ghb[HID + j])));
    float n = tanhf(gib[2 * HID + j] + r * ghb[2 * HID + j]);
    float hv = h[idx];
    float out = (1.0f - z) * n + z * hv;
    h[idx]    = out;
    h_bf[idx] = f2bf(out);
}

// ---------------- attention: scores, masked softmax, attended feat, lang_in ----------------
__global__ __launch_bounds__(256)
void attention_kernel(const float* __restrict__ feat_proj,     // [B*P, HID]
                      const float* __restrict__ hproj,         // [B, HID]
                      const float* __restrict__ w_att,         // [HID]
                      const int*   __restrict__ masks,         // [B, P]
                      const float* __restrict__ obj_feats,     // [B*P, FEAT]
                      const float* __restrict__ h1,            // [B, HID]
                      float* __restrict__ attn_out,            // + t*P ; idx b*T*P + p
                      unsigned short* __restrict__ lang_in_bf) // [B, LANG_K] bf16
{
    __shared__ float s_h[HID];
    __shared__ float s_w[HID];
    __shared__ float s_red[PP];
    const int b = blockIdx.x;
    const int t = threadIdx.x;

    for (int i = t; i < HID; i += PP) {
        s_h[i] = hproj[(size_t)b * HID + i];
        s_w[i] = w_att[i];
    }
    __syncthreads();

    const int p = t;
    const float* fp = feat_proj + ((size_t)b * PP + p) * HID;
    float score = 0.0f;
    for (int hh = 0; hh < HID; ++hh)
        score += tanhf(fp[hh] + s_h[hh]) * s_w[hh];
    if (masks[(size_t)b * PP + p] == 0) score = -1e30f;

    // masked softmax over the 256 proposals (block == P)
    s_red[t] = score; __syncthreads();
    for (int off = PP / 2; off > 0; off >>= 1) {
        if (t < off) s_red[t] = fmaxf(s_red[t], s_red[t + off]);
        __syncthreads();
    }
    float mx = s_red[0]; __syncthreads();
    float e = __expf(score - mx);
    s_red[t] = e; __syncthreads();
    for (int off = PP / 2; off > 0; off >>= 1) {
        if (t < off) s_red[t] += s_red[t + off];
        __syncthreads();
    }
    float a = e / s_red[0];
    __syncthreads();
    s_red[t] = a;
    attn_out[(size_t)b * (TT * PP) + p] = a;
    __syncthreads();

    // attended = sum_p attn[p] * obj_feats[b,p,:]
    if (t < FEAT) {
        float s = 0.0f;
        for (int p2 = 0; p2 < PP; ++p2)
            s += s_red[p2] * obj_feats[((size_t)b * PP + p2) * FEAT + t];
        lang_in_bf[(size_t)b * LANG_K + t] = f2bf(s);
    }
    for (int j = t; j < HID; j += PP)
        lang_in_bf[(size_t)b * LANG_K + FEAT + j] = f2bf(h1[(size_t)b * HID + j]);
}

// ---------------- host-side launch ----------------
static inline void launch_gemm(hipStream_t s, const unsigned short* A, int lda,
                               const unsigned short* Bw, int ldb, const float* bias,
                               float* C, int ldc, unsigned short* Cbf, int ldcb,
                               int M, int Nreal, int Npad, int Kpad, int act) {
    dim3 grid((Npad + 255) / 256, M / 16);
    wmma_gemm_bf16<<<grid, 128, 0, s>>>(A, lda, Bw, ldb, bias, C, ldc, Cbf, ldcb,
                                        M, Nreal, Npad, Kpad, act);
}

static inline void launch_pad(hipStream_t s, const float* src, int N, int K,
                              unsigned short* dst, int Npad, int Kpad) {
    int total = Npad * Kpad;
    pad_bf16_kernel<<<(total + 255) / 256, 256, 0, s>>>(src, N, K, dst, Npad, Kpad);
}

extern "C" void kernel_launch(void* const* d_in, const int* in_sizes, int n_in,
                              void* d_out, int out_size, void* d_ws, size_t ws_size,
                              hipStream_t stream) {
    // inputs (setup_inputs order)
    const float* word_embs = (const float*)d_in[0];
    const float* tfeat     = (const float*)d_in[1];
    const float* obj_feats = (const float*)d_in[2];
    const int*   masks     = (const int*)  d_in[3];
    const float* w_td   = (const float*)d_in[4];
    const float* b_td   = (const float*)d_in[5];
    const float* w_ih1  = (const float*)d_in[6];
    const float* w_hh1  = (const float*)d_in[7];
    const float* b_ih1  = (const float*)d_in[8];
    const float* b_hh1  = (const float*)d_in[9];
    const float* w_feat = (const float*)d_in[10];
    const float* w_hidd = (const float*)d_in[11];
    const float* w_att  = (const float*)d_in[12];
    const float* w_lang = (const float*)d_in[13];
    const float* b_lang = (const float*)d_in[14];
    const float* w_ih2  = (const float*)d_in[15];
    const float* w_hh2  = (const float*)d_in[16];
    const float* b_ih2  = (const float*)d_in[17];
    const float* b_hh2  = (const float*)d_in[18];
    const float* w_cls  = (const float*)d_in[19];
    const float* b_cls  = (const float*)d_in[20];

    // workspace bump allocator (256B aligned)
    char* wp = (char*)d_ws;
    auto alloc = [&](size_t bytes) -> void* {
        void* p = (void*)wp;
        wp += (bytes + 255) & ~(size_t)255;
        return p;
    };
    // bf16 padded weights
    unsigned short* w_td_p   = (unsigned short*)alloc((size_t)TD_N  * TD_K   * 2);
    unsigned short* w_ih1_p  = (unsigned short*)alloc((size_t)GRU_N * TD_LD  * 2);
    unsigned short* w_hh1_p  = (unsigned short*)alloc((size_t)GRU_N * HID    * 2);
    unsigned short* w_feat_p = (unsigned short*)alloc((size_t)HID   * FEAT   * 2);
    unsigned short* w_hidd_p = (unsigned short*)alloc((size_t)HID   * HID    * 2);
    unsigned short* w_lang_p = (unsigned short*)alloc((size_t)TD_N  * LANG_K * 2);
    unsigned short* w_ih2_p  = (unsigned short*)alloc((size_t)GRU_N * TD_LD  * 2);
    unsigned short* w_hh2_p  = (unsigned short*)alloc((size_t)GRU_N * HID    * 2);
    unsigned short* w_cls_p  = (unsigned short*)alloc((size_t)CLS_N * HID    * 2);
    unsigned short* obj_bf   = (unsigned short*)alloc((size_t)BB * PP * FEAT * 2);
    // big f32 scratch
    float* feat_proj = (float*)alloc((size_t)BB * PP * HID * 4);   // 64 MB
    // activation scratch (zeroed each launch: h state + all pad columns)
    char* zbase = wp;
    unsigned short* td_in_bf   = (unsigned short*)alloc((size_t)BB * TD_K   * 2);
    unsigned short* td_bf      = (unsigned short*)alloc((size_t)BB * TD_LD  * 2);
    unsigned short* lang_bf    = (unsigned short*)alloc((size_t)BB * TD_LD  * 2);
    unsigned short* lang_in_bf = (unsigned short*)alloc((size_t)BB * LANG_K * 2);
    unsigned short* h1_bf      = (unsigned short*)alloc((size_t)BB * HID    * 2);
    unsigned short* h2_bf      = (unsigned short*)alloc((size_t)BB * HID    * 2);
    float* gi    = (float*)alloc((size_t)BB * GRU_N * 4);
    float* gh    = (float*)alloc((size_t)BB * GRU_N * 4);
    float* h1    = (float*)alloc((size_t)BB * HID   * 4);
    float* h2    = (float*)alloc((size_t)BB * HID   * 4);
    float* hproj = (float*)alloc((size_t)BB * HID   * 4);
    int zero_words = (int)((wp - zbase) / 4);

    float* logits_out = (float*)d_out;                       // [B, T, V]
    float* attn_out   = logits_out + (size_t)BB * TT * VV;   // [B, T, P]

    // ---- weight + obj_feats conversion (bf16, padded) ----
    launch_pad(stream, w_td,   EMB,   940,    w_td_p,   TD_N,  TD_K);
    launch_pad(stream, w_ih1,  GRU_N, EMB,    w_ih1_p,  GRU_N, TD_LD);
    launch_pad(stream, w_hh1,  GRU_N, HID,    w_hh1_p,  GRU_N, HID);
    launch_pad(stream, w_feat, HID,   FEAT,   w_feat_p, HID,   FEAT);
    launch_pad(stream, w_hidd, HID,   HID,    w_hidd_p, HID,   HID);
    launch_pad(stream, w_lang, EMB,   LANG_K, w_lang_p, TD_N,  LANG_K);
    launch_pad(stream, w_ih2,  GRU_N, EMB,    w_ih2_p,  GRU_N, TD_LD);
    launch_pad(stream, w_hh2,  GRU_N, HID,    w_hh2_p,  GRU_N, HID);
    launch_pad(stream, w_cls,  VV,    HID,    w_cls_p,  CLS_N, HID);
    launch_pad(stream, obj_feats, BB * PP, FEAT, obj_bf, BB * PP, FEAT);

    // ---- zero activation state (h1/h2 + pad columns stay 0 forever) ----
    zero_u32_kernel<<<(zero_words + 255) / 256, 256, 0, stream>>>(
        (unsigned int*)zbase, zero_words);

    // ---- feat_proj = obj_feats @ w_feat^T : [B*P, HID] ----
    launch_gemm(stream, obj_bf, FEAT, w_feat_p, FEAT, nullptr,
                feat_proj, HID, nullptr, 0, BB * PP, HID, HID, FEAT, 0);

    // ---- timestep loop ----
    for (int t = 0; t < TT; ++t) {
        // td_in = [x_t, h2, target_feat]  (bf16)
        concat_td_kernel<<<(BB * 940 + 255) / 256, 256, 0, stream>>>(
            word_embs, t, h2, tfeat, td_in_bf);
        // td = relu(td_in @ w_td^T + b_td)  -> bf16 only
        launch_gemm(stream, td_in_bf, TD_K, w_td_p, TD_K, b_td,
                    nullptr, 0, td_bf, TD_LD, BB, EMB, TD_N, TD_K, 1);
        // GRU1
        launch_gemm(stream, td_bf, TD_LD, w_ih1_p, TD_LD, b_ih1,
                    gi, GRU_N, nullptr, 0, BB, GRU_N, GRU_N, TD_LD, 0);
        launch_gemm(stream, h1_bf, HID, w_hh1_p, HID, b_hh1,
                    gh, GRU_N, nullptr, 0, BB, GRU_N, GRU_N, HID, 0);
        gru_gate_kernel<<<(BB * HID + 255) / 256, 256, 0, stream>>>(gi, gh, h1, h1_bf);
        // hproj = h1 @ w_hidd^T
        launch_gemm(stream, h1_bf, HID, w_hidd_p, HID, nullptr,
                    hproj, HID, nullptr, 0, BB, HID, HID, HID, 0);
        // attention + lang_in build (bf16)
        attention_kernel<<<BB, PP, 0, stream>>>(
            feat_proj, hproj, w_att, masks, obj_feats, h1,
            attn_out + (size_t)t * PP, lang_in_bf);
        // lang = relu(lang_in @ w_lang^T + b_lang)  -> bf16 only
        launch_gemm(stream, lang_in_bf, LANG_K, w_lang_p, LANG_K, b_lang,
                    nullptr, 0, lang_bf, TD_LD, BB, EMB, TD_N, LANG_K, 1);
        // GRU2
        launch_gemm(stream, lang_bf, TD_LD, w_ih2_p, TD_LD, b_ih2,
                    gi, GRU_N, nullptr, 0, BB, GRU_N, GRU_N, TD_LD, 0);
        launch_gemm(stream, h2_bf, HID, w_hh2_p, HID, b_hh2,
                    gh, GRU_N, nullptr, 0, BB, GRU_N, GRU_N, HID, 0);
        gru_gate_kernel<<<(BB * HID + 255) / 256, 256, 0, stream>>>(gi, gh, h2, h2_bf);
        // logits[:, t, :] = h2 @ w_cls^T + b_cls  (strided write into d_out)
        launch_gemm(stream, h2_bf, HID, w_cls_p, HID, b_cls,
                    logits_out + (size_t)t * VV, TT * VV, nullptr, 0,
                    BB, VV, CLS_N, HID, 0);
    }
    (void)in_sizes; (void)n_in; (void)out_size; (void)ws_size;
}
